// MAB_55087250539003
// MI455X (gfx1250) — matrix operand order repaired and checked by reading, not compiled, not run on hardware
//
#include <hip/hip_runtime.h>
#include <hip/hip_bf16.h>
#include <math.h>

// ---------------------------------------------------------------------------
// MHA with tanh-clipped scores, masked fill, log-softmax-as-weights.
// All GEMMs run on v_wmma_f32_16x16x32_bf16 (f32 accumulate).
// out_head = S@V - lse * colsum(V)   (since attn = S - lse row-wise)
// ---------------------------------------------------------------------------

typedef __attribute__((ext_vector_type(16))) __bf16 v16bf;
typedef __attribute__((ext_vector_type(8)))  float  v8f;
typedef __attribute__((ext_vector_type(4)))  float  f4;

#define D_MODEL 512
#define LSEQ    2048
#define NHEADS  2
#define DHEAD   64
#define BATCH   4
#define BH      (BATCH * NHEADS)

__device__ inline v8f wmma_bf16(v16bf a, v16bf b, v8f c) {
  return __builtin_amdgcn_wmma_f32_16x16x32_bf16(false, a, false, b,
                                                 (short)0, c, false, false);
}

// 8 bf16 from p0 (elements 0..7) + 8 bf16 from p1 (elements 8..15).
// Each chunk is 16B; pointers must be 16B aligned.
__device__ inline v16bf ld_bf16_8_8(const __bf16* p0, const __bf16* p1) {
  union U { f4 f[2]; v16bf v; } u;
  u.f[0] = *(const f4*)p0;
  u.f[1] = *(const f4*)p1;
  return u.v;
}

// 8 f32 from p0 + 8 f32 from p1, converted to bf16 (A-fragment build).
__device__ inline v16bf ld_f32_cvt(const float* p0, const float* p1) {
  f4 a0 = ((const f4*)p0)[0], a1 = ((const f4*)p0)[1];
  f4 b0 = ((const f4*)p1)[0], b1 = ((const f4*)p1)[1];
  v16bf v;
#pragma unroll
  for (int i = 0; i < 4; ++i) {
    v[i]      = (__bf16)a0[i];
    v[4 + i]  = (__bf16)a1[i];
    v[8 + i]  = (__bf16)b0[i];
    v[12 + i] = (__bf16)b1[i];
  }
  return v;
}

// ---------------------------------------------------------------------------
// Weight prep: transpose to k-major and convert to bf16.
// Wt*[n][k] = W*[k][n]  (n<128,k<512);  Wot[e][c] = Wo[c][e] (e<512,c<128)
// ---------------------------------------------------------------------------
__global__ __launch_bounds__(256)
void k_prep(const float* __restrict__ Wq, const float* __restrict__ Wk,
            const float* __restrict__ Wv, const float* __restrict__ Wo,
            __bf16* __restrict__ Wtq, __bf16* __restrict__ Wtk,
            __bf16* __restrict__ Wtv, __bf16* __restrict__ Wot) {
  int i = blockIdx.x * blockDim.x + threadIdx.x;   // 0..65535
  {
    int n = i >> 9, k = i & 511;                   // 128 x 512
    Wtq[i] = (__bf16)Wq[k * 128 + n];
    Wtk[i] = (__bf16)Wk[k * 128 + n];
    Wtv[i] = (__bf16)Wv[k * 128 + n];
  }
  {
    int e = i >> 7, c = i & 127;                   // 512 x 128
    Wot[i] = (__bf16)Wo[c * 512 + e];
  }
}

// ---------------------------------------------------------------------------
// Projection: [8192 x 512] @ [512 x 128] + bias, head-split bf16 output.
// vmode==0 -> dst[bh][l][d] (q,k)   vmode==1 -> dst[bh][d][l] (v transposed)
// ---------------------------------------------------------------------------
__global__ __launch_bounds__(32)
void k_proj(const float* __restrict__ X, const __bf16* __restrict__ Wt,
            const float* __restrict__ bias, __bf16* __restrict__ dst,
            int vmode) {
  int mt = blockIdx.x;                 // 0..511 (row tile)
  int ct = blockIdx.y;                 // 0..7   (col tile of 128)
  int l = threadIdx.x, c16 = l & 15, g = l >> 4;
  int off = g * 8, ktop = g * 16;
  int m = mt * 16 + c16;
  int n = ct * 16 + c16;
  const float*  xrow = X  + (size_t)m * D_MODEL;
  const __bf16* wrow = Wt + (size_t)n * D_MODEL;

  v8f acc = {};
#pragma unroll 4
  for (int kk = 0; kk < D_MODEL; kk += 32) {
    v16bf a = ld_f32_cvt(xrow + kk + off, xrow + kk + off + 16);
    v16bf b = ld_bf16_8_8(wrow + kk + off, wrow + kk + off + 16);
    acc = wmma_bf16(a, b, acc);
  }

  float bv = bias[n];
  int h = n >> 6, d = n & 63;
#pragma unroll
  for (int r = 0; r < 8; ++r) {
    int row = mt * 16 + r + off;       // global row in [0,8192)
    int bb = row >> 11, lr = row & 2047;
    float val = acc[r] + bv;
    if (!vmode)
      dst[((size_t)(bb * NHEADS + h) * LSEQ + lr) * DHEAD + d] = (__bf16)val;
    else
      dst[((size_t)(bb * NHEADS + h) * DHEAD + d) * LSEQ + lr] = (__bf16)val;
  }
}

// ---------------------------------------------------------------------------
// Column sums of V per head: vcol[bh*64+d] = sum_n vt[bh][d][n]
// ---------------------------------------------------------------------------
__global__ __launch_bounds__(32)
void k_vsum(const __bf16* __restrict__ vt, float* __restrict__ vcol) {
  __shared__ float red[32];
  int row = blockIdx.x;                              // 0..511
  const __bf16* p = vt + (size_t)row * LSEQ;
  float s = 0.f;
  for (int n = threadIdx.x; n < LSEQ; n += 32) s += (float)p[n];
  red[threadIdx.x] = s;
  __syncthreads();
  if (threadIdx.x == 0) {
    float t = 0.f;
    for (int j = 0; j < 32; ++j) t += red[j];
    vcol[row] = t;
  }
}

// ---------------------------------------------------------------------------
// Attention core: one wave per (bh, 16-row q tile).
// LDS strip holds the full 16 x 2048 f32 score row-block (128 KB).
// ---------------------------------------------------------------------------
__global__ __launch_bounds__(32)
void k_attn(const __bf16* __restrict__ qh, const __bf16* __restrict__ kh,
            const __bf16* __restrict__ vt, const unsigned char* __restrict__ mask,
            const float* __restrict__ vcol, float* __restrict__ out_head,
            float* __restrict__ attn_out) {
  extern __shared__ float smem[];
  float* strip = smem;                 // 16 * 2048
  float* red   = smem + 16 * LSEQ;     // 32 * 8
  float* lseS  = red + 256;            // 16

  int mt = blockIdx.x;                 // q tile 0..127
  int bh = blockIdx.y;                 // 0..7
  int b  = bh >> 1;
  int l = threadIdx.x, c16 = l & 15, g = l >> 4;
  int off = g * 8, ktop = g * 16;
  int m = mt * 16 + c16;

  // Q A-fragments for this tile (16x64 = two 16x32 chunks), kept in VGPRs.
  const __bf16* qrow = qh + ((size_t)bh * LSEQ + m) * DHEAD;
  v16bf aq0 = ld_bf16_8_8(qrow + off,      qrow + off + 16);
  v16bf aq1 = ld_bf16_8_8(qrow + 32 + off, qrow + 32 + off + 16);

  const __bf16* kbase = kh + (size_t)bh * LSEQ * DHEAD;
  const __bf16* vbase = vt + (size_t)bh * DHEAD * LSEQ;
  const unsigned char* mrow = mask + (size_t)b * LSEQ;

  float accSum[8];
#pragma unroll
  for (int r = 0; r < 8; ++r) accSum[r] = 0.f;
  v8f acc_o[4] = {v8f{}, v8f{}, v8f{}, v8f{}};

  for (int nt = 0; nt < LSEQ / 16; ++nt) {
    int n0 = nt * 16;
    const __bf16* krow = kbase + (size_t)(n0 + c16) * DHEAD;
    __builtin_prefetch(krow + 16 * DHEAD, 0, 1);   // next k tile

    v8f s = {};
    s = wmma_bf16(aq0, ld_bf16_8_8(krow + off,      krow + off + 16),      s);
    s = wmma_bf16(aq1, ld_bf16_8_8(krow + 32 + off, krow + 32 + off + 16), s);

    unsigned char mb = mrow[n0 + c16];             // this lane's column
#pragma unroll
    for (int r = 0; r < 8; ++r) {
      float sv = s[r] * 0.125f;                    // 1/sqrt(64)
      sv = 10.0f * tanhf(sv);
      if (mb) sv = -10.0f;
      strip[(size_t)(r + off) * LSEQ + n0 + c16] = sv;
      accSum[r] += __expf(sv);
    }
    __syncthreads();

    if (nt & 1) {                                  // 32 score columns ready
      int p0 = (nt - 1) * 16;
      const float* srow = strip + (size_t)c16 * LSEQ + p0;
      v16bf as = ld_f32_cvt(srow + off, srow + off + 16);
#pragma unroll
      for (int t = 0; t < 4; ++t) {
        const __bf16* vrow = vbase + (size_t)(t * 16 + c16) * LSEQ + p0 + off;
        acc_o[t] = wmma_bf16(as, ld_bf16_8_8(vrow, vrow + 16), acc_o[t]);
      }
    }
  }

  // Row-wise log-sum-exp via LDS reduction across the 16 lanes of each group.
#pragma unroll
  for (int r = 0; r < 8; ++r) red[l * 8 + r] = accSum[r];
  __syncthreads();
  if (l < 16) {
    int rr = l & 7, gg = (l >> 3) * 16;
    float t = 0.f;
    for (int j = 0; j < 16; ++j) t += red[(gg + j) * 8 + rr];
    lseS[l] = __logf(t);
  }
  __syncthreads();

  // out_head = S@V - lse * colsum(V)
#pragma unroll
  for (int t = 0; t < 4; ++t) {
    int d = t * 16 + c16;
    float vc = vcol[bh * DHEAD + d];
#pragma unroll
    for (int r = 0; r < 8; ++r) {
      int row = r + off;
      float val = acc_o[t][r] - lseS[row] * vc;
      out_head[((size_t)bh * LSEQ + mt * 16 + row) * DHEAD + d] = val;
    }
  }

  // attn = strip - lse, streamed out with float4 stores.
  float* abase = attn_out + ((size_t)bh * LSEQ + (size_t)mt * 16) * LSEQ;
  for (int idx = l; idx < 16 * (LSEQ / 4); idx += 32) {
    int row = idx >> 9;
    int cq  = (idx & 511) * 4;
    f4 v4 = *(const f4*)&strip[(size_t)row * LSEQ + cq];
    float ls = lseS[row];
#pragma unroll
    for (int j = 0; j < 4; ++j) v4[j] -= ls;
    *(f4*)&abase[(size_t)row * LSEQ + cq] = v4;
  }
}

// ---------------------------------------------------------------------------
// Output projection: [8192 x 128] @ [128 x 512] + bo (heads re-interleaved).
// ---------------------------------------------------------------------------
__global__ __launch_bounds__(32)
void k_oproj(const float* __restrict__ oh, const __bf16* __restrict__ Wot,
             const float* __restrict__ bo, float* __restrict__ out) {
  int mt = blockIdx.x;                 // 0..511
  int et = blockIdx.y;                 // 0..31
  int l = threadIdx.x, c16 = l & 15, g = l >> 4;
  int off = g * 8, ktop = g * 16;
  int m = mt * 16 + c16;
  int bb = m >> 11, lr = m & 2047;
  int e = et * 16 + c16;
  const __bf16* wrow = Wot + (size_t)e * 128;

  v8f acc = {};
#pragma unroll
  for (int kk = 0; kk < 128; kk += 32) {
    int c0 = kk + off, c1 = kk + off + 16;   // 8-contig chunks, never cross head
    const float* p0 =
        oh + ((size_t)(bb * NHEADS + (c0 >> 6)) * LSEQ + lr) * DHEAD + (c0 & 63);
    const float* p1 =
        oh + ((size_t)(bb * NHEADS + (c1 >> 6)) * LSEQ + lr) * DHEAD + (c1 & 63);
    v16bf a = ld_f32_cvt(p0, p1);
    v16bf b = ld_bf16_8_8(wrow + kk + off, wrow + kk + off + 16);
    acc = wmma_bf16(a, b, acc);
  }

  float bv = bo[e];
#pragma unroll
  for (int r = 0; r < 8; ++r) {
    int row = mt * 16 + r + off;
    out[(size_t)row * D_MODEL + e] = acc[r] + bv;
  }
}

// ---------------------------------------------------------------------------
extern "C" void kernel_launch(void* const* d_in, const int* in_sizes, int n_in,
                              void* d_out, int out_size, void* d_ws, size_t ws_size,
                              hipStream_t stream) {
  const float* Q  = (const float*)d_in[0];
  const float* K  = (const float*)d_in[1];
  const float* V  = (const float*)d_in[2];
  const unsigned char* mask = (const unsigned char*)d_in[3];  // jax bool = 1B
  const float* Wq = (const float*)d_in[4];
  const float* bq = (const float*)d_in[5];
  const float* Wk = (const float*)d_in[6];
  const float* bk = (const float*)d_in[7];
  const float* Wv = (const float*)d_in[8];
  const float* bv = (const float*)d_in[9];
  const float* Wo = (const float*)d_in[10];
  const float* bo = (const float*)d_in[11];

  float* out  = (float*)d_out;                       // [4,2048,512]
  float* attn = out + (size_t)BATCH * LSEQ * D_MODEL;// [4,2,2048,2048]

  char* ws = (char*)d_ws;
  __bf16* qh  = (__bf16*)(ws);                       // [8][2048][64]  2 MB
  __bf16* kh  = (__bf16*)(ws + (2u << 20));          // [8][2048][64]  2 MB
  __bf16* vt  = (__bf16*)(ws + (4u << 20));          // [8][64][2048]  2 MB
  float*  oh  = (float* )(ws + (6u << 20));          // [8][2048][64]  4 MB
  __bf16* Wtq = (__bf16*)(ws + (10u << 20));         // [128][512]
  __bf16* Wtk = Wtq + 128 * 512;
  __bf16* Wtv = Wtk + 128 * 512;
  __bf16* Wot = Wtv + 128 * 512;                     // [512][128]
  float*  vcol = (float*)(ws + (10u << 20) + 4u * 128 * 512 * 2);  // [512]

  k_prep<<<256, 256, 0, stream>>>(Wq, Wk, Wv, Wo, Wtq, Wtk, Wtv, Wot);

  dim3 g1(512, 8);
  k_proj<<<g1, 32, 0, stream>>>(Q, Wtq, bq, qh, 0);
  k_proj<<<g1, 32, 0, stream>>>(K, Wtk, bk, kh, 0);
  k_proj<<<g1, 32, 0, stream>>>(V, Wtv, bv, vt, 1);

  k_vsum<<<512, 32, 0, stream>>>(vt, vcol);

  dim3 g2(128, 8);
  size_t smem = (size_t)(16 * LSEQ + 32 * 8 + 16) * sizeof(float); // ~129 KB
  k_attn<<<g2, 32, smem, stream>>>(qh, kh, vt, mask, vcol, oh, attn);

  dim3 g3(512, 32);
  k_oproj<<<g3, 32, 0, stream>>>(oh, Wot, bo, out);
}
